// SymGAT_processor_89601607729413
// MI455X (gfx1250) — matrix operand order repaired and checked by reading, not compiled, not run on hardware
//
#include <hip/hip_runtime.h>

// ---------------------------------------------------------------------------
// SymGAT on MI455X (gfx1250): all E×F×F and N×F×F GEMMs via
// v_wmma_f32_16x16x32_f16 (fp32 accumulate), fp16 shadow activations for
// L2-resident gathers, fused BN-stat + attention-logit epilogues, atomic
// segment softmax / scatter aggregation. B-fragments are register
// double-buffered; sched_group_barrier emits a phase-unambiguous pipelined
// group sequence so step k+1's weight loads issue between step k-1's and
// step k's WMMA groups.
// ---------------------------------------------------------------------------

typedef _Float16 v16h __attribute__((ext_vector_type(16)));
typedef float    v8f  __attribute__((ext_vector_type(8)));

constexpr int LYR = 4;
constexpr int NN  = 50000;
constexpr int EE  = 800000;
constexpr int FD  = 256;
constexpr int BM  = 32;          // rows (edges/nodes) per block tile
constexpr float NEG_SLOPE = 0.2f;

union AB16 { v16h v; uint4 q[2]; };
struct h4v { _Float16 a, b, c, d; };

struct EdgeSmem {
    union U { _Float16 a[3][BM][FD]; float c[BM][FD]; } u;   // 48 KB / 32 KB
    int idx[2][BM];
};
struct TileSmem {
    union U { _Float16 a[BM][FD]; float c[BM][FD]; } u;      // 16 KB / 32 KB
};

// ---- float atomic max via integer punning (works for any sign) -------------
__device__ inline void atomicMaxF(float* addr, float v) {
    if (v >= 0.0f) atomicMax((int*)addr, __float_as_int(v));
    else           atomicMin((unsigned int*)addr, (unsigned int)__float_as_int(v));
}

// ---- wave-level 16(M)x64(N) WMMA strip over K=256 --------------------------
// A layout (16-bit 16x32): lanes 0-15 row m hold K[k0..k0+15];
// lanes 16-31 hold K[k0+16..k0+31]  -> two 16B LDS loads per step.
// B layout (16-bit 32x16): lane n%16 holds 16 consecutive K halves starting at
// k0 + (lane>=16)*16 -> contiguous loads from the pre-transposed f16 weight.
// Software pipeline (group-barrier enforced program order):
//   [vmem x8: B0] then per step k: [ds x2: Ak][vmem x8: B(k+1)][wmma x4: k]
// Group totals exactly cover the region (64 vmem, 16 ds, 32 wmma), so the
// only dependency-legal assignment is the pipelined one: B(k) completes one
// full group-cycle before its consuming WMMAs.
__device__ inline void mma_accum(const _Float16 (*A)[FD], const _Float16* WT,
                                 int wm, int wn, int lane, v8f acc[4]) {
    const int l16 = lane & 15, lhalf = lane >> 4;
    const _Float16* wbase = WT + (size_t)(wn * 64 + l16) * FD + lhalf * 16;
    const _Float16* arow  = A[wm * 16 + l16] + lhalf * 16;

    AB16 b[2][4];
#pragma unroll
    for (int t = 0; t < 4; ++t) {
        const _Float16* bp = wbase + t * 16 * FD;
        b[0][t].q[0] = *(const uint4*)bp;
        b[0][t].q[1] = *(const uint4*)(bp + 8);
    }
    __builtin_amdgcn_sched_group_barrier(0x020, 8, 0);      // B0 preload
#pragma unroll
    for (int ks = 0; ks < 8; ++ks) {
        const int cur = ks & 1;
        if (ks < 7) {
#pragma unroll
            for (int t = 0; t < 4; ++t) {
                const _Float16* bp = wbase + t * 16 * FD + (ks + 1) * 32;
                b[cur ^ 1][t].q[0] = *(const uint4*)bp;
                b[cur ^ 1][t].q[1] = *(const uint4*)(bp + 8);
            }
        }
        AB16 a;
        a.q[0] = *(const uint4*)(arow + ks * 32);
        a.q[1] = *(const uint4*)(arow + ks * 32 + 8);
#pragma unroll
        for (int t = 0; t < 4; ++t)
            acc[t] = __builtin_amdgcn_wmma_f32_16x16x32_f16(
                false, a.v, false, b[cur][t].v, (short)0, acc[t], false, false);
        // phase-unambiguous pipeline: A(k), then B(k+1), then WMMA(k)
        __builtin_amdgcn_sched_group_barrier(0x100, 2, 0);  // 2 DS reads (A k)
        if (ks < 7)
            __builtin_amdgcn_sched_group_barrier(0x020, 8, 0);  // 8 VMEM (B k+1)
        __builtin_amdgcn_sched_group_barrier(0x008, 4, 0);  // 4 WMMA (k)
    }
}

// C layout: VGPR r -> row = (lane>>4)*8 + r, col = lane&15 within 16x16 tile.
__device__ inline void store_acc(float (*C)[FD], int wm, int wn, int lane,
                                 const v8f acc[4]) {
    const int l16 = lane & 15, lhalf = lane >> 4;
#pragma unroll
    for (int t = 0; t < 4; ++t)
#pragma unroll
        for (int r = 0; r < 8; ++r)
            C[wm * 16 + lhalf * 8 + r][wn * 64 + t * 16 + l16] = acc[t][r];
}

// ---------------------------------------------------------------------------
// Kernel 1: e_tmp = e@We + h_s@Ws + h_d@Wd  (+ fused epilogue:
//   spill e_tmp fp32, attention logits = leaky(e_tmp . a), BN partial sums)
// ---------------------------------------------------------------------------
__global__ __launch_bounds__(256)
void k_edge_gemm(const _Float16* __restrict__ e16, const _Float16* __restrict__ h16,
                 const int* __restrict__ src, const int* __restrict__ dst,
                 const _Float16* __restrict__ wt,            // [6][F][F] transposed f16
                 const float* __restrict__ af, const float* __restrict__ ab,
                 float* __restrict__ e_tmp,
                 float* __restrict__ logit_f, float* __restrict__ logit_b,
                 float* __restrict__ bn_sum, float* __restrict__ bn_ss) {
    __shared__ EdgeSmem sm;
    const int tid = threadIdx.x;
    const int eBase = blockIdx.x * BM;

    if (tid < BM)          sm.idx[0][tid]      = src[eBase + tid];
    else if (tid < 2 * BM) sm.idx[1][tid - BM] = dst[eBase + tid - BM];
    __syncthreads();

    // stage A tiles: [e | h_src | h_dst] as f16, 32 rows x 256 (48 KB)
    for (int i = tid; i < 3 * BM * (FD / 8); i += 256) {
        int mat = i / (BM * (FD / 8));
        int rem = i % (BM * (FD / 8));
        int r = rem / (FD / 8), ch = rem % (FD / 8);
        const _Float16* p = (mat == 0)
            ? e16 + (size_t)(eBase + r) * FD
            : h16 + (size_t)sm.idx[mat - 1][r] * FD;
        *(uint4*)&sm.u.a[mat][r][ch * 8] = *(const uint4*)(p + ch * 8);
    }
    __syncthreads();

    const int lane = tid & 31, wave = tid >> 5;
    const int wm = wave >> 2, wn = wave & 3;

    v8f acc[4] = {};
    for (int mat = 0; mat < 3; ++mat)
        mma_accum(sm.u.a[mat], wt + ((size_t)mat << 16), wm, wn, lane, acc);

    __syncthreads();                      // done with f16 staging memory
    store_acc(sm.u.c, wm, wn, lane, acc);
    __syncthreads();

    // (1) spill e_tmp fp32 (coalesced float4)
    {
        float4* g4 = (float4*)(e_tmp + (size_t)eBase * FD);
        const float4* s4 = (const float4*)sm.u.c;
        for (int i = tid; i < BM * FD / 4; i += 256) g4[i] = s4[i];
    }
    // (2) attention logits: 8 threads per edge row, shuffle-reduce width 8
    {
        const int row = tid >> 3, part = tid & 7;
        const float* crow = sm.u.c[row];
        float pf = 0.f, pb = 0.f;
        for (int j = 0; j < 32; ++j) {
            float x = crow[part * 32 + j];
            pf += x * af[part * 32 + j];
            pb += x * ab[part * 32 + j];
        }
        for (int d = 4; d; d >>= 1) {
            pf += __shfl_down(pf, d, 8);
            pb += __shfl_down(pb, d, 8);
        }
        if (part == 0) {
            logit_f[eBase + row] = pf > 0.f ? pf : pf * NEG_SLOPE;
            logit_b[eBase + row] = pb > 0.f ? pb : pb * NEG_SLOPE;
        }
    }
    // (3) BN partial sums: one feature column per thread
    {
        float s = 0.f, ss = 0.f;
        for (int r = 0; r < BM; ++r) {
            float x = sm.u.c[r][tid];
            s += x; ss += x * x;
        }
        atomicAdd(&bn_sum[tid], s);
        atomicAdd(&bn_ss[tid], ss);
    }
}

// ---------------------------------------------------------------------------
// Kernel 2: agg += alpha[edge] * (h[gat[edge]] @ W), scattered by seg[edge]
// (recomputing h@W here is cheaper than spilling another E*F fp32 tensor)
// ---------------------------------------------------------------------------
__global__ __launch_bounds__(256)
void k_agg_gemm(const _Float16* __restrict__ h16,
                const int* __restrict__ gat, const int* __restrict__ seg,
                const _Float16* __restrict__ WT,
                const float* __restrict__ ex, const float* __restrict__ den,
                float* __restrict__ agg) {
    __shared__ TileSmem sm;
    __shared__ float sAlpha[BM];
    __shared__ int sSeg[BM], sGat[BM];
    const int tid = threadIdx.x;
    const int eBase = blockIdx.x * BM;

    if (tid < BM) {
        int eid = eBase + tid;
        sGat[tid] = gat[eid];
        int sg = seg[eid];
        sSeg[tid] = sg;
        sAlpha[tid] = ex[eid] / (den[sg] + 1e-9f);
    }
    __syncthreads();

    for (int i = tid; i < BM * (FD / 8); i += 256) {
        int r = i / (FD / 8), ch = i % (FD / 8);
        *(uint4*)&sm.u.a[r][ch * 8] =
            *(const uint4*)(h16 + (size_t)sGat[r] * FD + ch * 8);
    }
    __syncthreads();

    const int lane = tid & 31, wave = tid >> 5;
    const int wm = wave >> 2, wn = wave & 3;
    v8f acc[4] = {};
    mma_accum(sm.u.a, WT, wm, wn, lane, acc);
    __syncthreads();
    store_acc(sm.u.c, wm, wn, lane, acc);
    __syncthreads();

    for (int i = tid; i < BM * FD; i += 256) {
        int r = i >> 8, c = i & 255;
        atomicAdd(&agg[(size_t)sSeg[r] * FD + c], sAlpha[r] * sm.u.c[r][c]);
    }
}

// ---------------------------------------------------------------------------
// Kernel 3: h_tmp = h@Wself + agg_f + agg_b  (+ BN partial sums)
// ---------------------------------------------------------------------------
__global__ __launch_bounds__(256)
void k_node_gemm(const _Float16* __restrict__ h16, const _Float16* __restrict__ WT,
                 const float* __restrict__ agg_f, const float* __restrict__ agg_b,
                 float* __restrict__ h_tmp,
                 float* __restrict__ bn_sum, float* __restrict__ bn_ss) {
    __shared__ TileSmem sm;
    const int tid = threadIdx.x;
    const int base = blockIdx.x * BM;

    for (int i = tid; i < BM * (FD / 8); i += 256) {
        int r = i / (FD / 8), ch = i % (FD / 8);
        int node = base + r;
        uint4 v = make_uint4(0u, 0u, 0u, 0u);
        if (node < NN) v = *(const uint4*)(h16 + (size_t)node * FD + ch * 8);
        *(uint4*)&sm.u.a[r][ch * 8] = v;
    }
    __syncthreads();

    const int lane = tid & 31, wave = tid >> 5;
    const int wm = wave >> 2, wn = wave & 3;
    v8f acc[4] = {};
    mma_accum(sm.u.a, WT, wm, wn, lane, acc);
    __syncthreads();
    store_acc(sm.u.c, wm, wn, lane, acc);
    __syncthreads();

    for (int i = tid; i < BM * FD; i += 256) {
        int r = i >> 8, c = i & 255;
        int node = base + r;
        if (node < NN) {
            size_t off = (size_t)node * FD + c;
            float v = sm.u.c[r][c] + agg_f[off] + agg_b[off];
            sm.u.c[r][c] = v;
            h_tmp[off] = v;
        }
    }
    __syncthreads();

    float s = 0.f, ss = 0.f;
    for (int r = 0; r < BM; ++r)
        if (base + r < NN) {
            float x = sm.u.c[r][tid];
            s += x; ss += x * x;
        }
    atomicAdd(&bn_sum[tid], s);
    atomicAdd(&bn_ss[tid], ss);
}

// ---------------------------------------------------------------------------
// Small kernels
// ---------------------------------------------------------------------------
__global__ void k_segmax(const float* __restrict__ lf, const float* __restrict__ lb,
                         const int* __restrict__ src, const int* __restrict__ dst,
                         float* __restrict__ mxf, float* __restrict__ mxb) {
    int i = blockIdx.x * 256 + threadIdx.x;
    if (i < EE) {
        atomicMaxF(&mxf[dst[i]], lf[i]);
        atomicMaxF(&mxb[src[i]], lb[i]);
    }
}

__global__ void k_expsum(float* __restrict__ lf, float* __restrict__ lb,
                         const int* __restrict__ src, const int* __restrict__ dst,
                         const float* __restrict__ mxf, const float* __restrict__ mxb,
                         float* __restrict__ denf, float* __restrict__ denb) {
    int i = blockIdx.x * 256 + threadIdx.x;
    if (i < EE) {
        int d = dst[i], s = src[i];
        float ef = expf(lf[i] - mxf[d]);
        float eb = expf(lb[i] - mxb[s]);
        lf[i] = ef; lb[i] = eb;
        atomicAdd(&denf[d], ef);
        atomicAdd(&denb[s], eb);
    }
}

__global__ void k_bn_finalize(const float* __restrict__ sum, const float* __restrict__ ss,
                              float* __restrict__ mu, float* __restrict__ rs,
                              float inv_count) {
    int c = threadIdx.x;
    float m = sum[c] * inv_count;
    float v = ss[c] * inv_count - m * m;
    mu[c] = m;
    rs[c] = rsqrtf(v + 1e-5f);
}

// out = residual + relu(bn(tmp)); writes fp32 (in place) + f16 shadow
__global__ void k_update(const float* __restrict__ tmp,
                         const float* __restrict__ mu, const float* __restrict__ rs,
                         const float* __restrict__ g, const float* __restrict__ b,
                         float* __restrict__ io, _Float16* __restrict__ out16,
                         size_t n4) {
    size_t i = (size_t)blockIdx.x * 256 + threadIdx.x;
    if (i >= n4) return;
    float4 x = ((const float4*)tmp)[i];
    float4 r0 = ((const float4*)io)[i];
    int c = (int)((i << 2) & (FD - 1));
    float o0 = r0.x + fmaxf((x.x - mu[c + 0]) * rs[c + 0] * g[c + 0] + b[c + 0], 0.f);
    float o1 = r0.y + fmaxf((x.y - mu[c + 1]) * rs[c + 1] * g[c + 1] + b[c + 1], 0.f);
    float o2 = r0.z + fmaxf((x.z - mu[c + 2]) * rs[c + 2] * g[c + 2] + b[c + 2], 0.f);
    float o3 = r0.w + fmaxf((x.w - mu[c + 3]) * rs[c + 3] * g[c + 3] + b[c + 3], 0.f);
    ((float4*)io)[i] = make_float4(o0, o1, o2, o3);
    h4v hv = { (_Float16)o0, (_Float16)o1, (_Float16)o2, (_Float16)o3 };
    ((h4v*)out16)[i] = hv;
}

__global__ void k_fill(float* __restrict__ p, float v, size_t n) {
    size_t i = (size_t)blockIdx.x * 256 + threadIdx.x;
    if (i < n) p[i] = v;
}

__global__ void k_cvt(const float* __restrict__ x, _Float16* __restrict__ y, size_t n) {
    size_t i = (size_t)blockIdx.x * 256 + threadIdx.x;
    if (i < n) y[i] = (_Float16)x[i];
}

// convert layer weights to f16, transposed (wt[m][n][k] = W[m][k][n])
__global__ void k_prep_w(const float* __restrict__ We, const float* __restrict__ Ws,
                         const float* __restrict__ Wd, const float* __restrict__ Wself,
                         const float* __restrict__ Wf, const float* __restrict__ Wb,
                         _Float16* __restrict__ wt) {
    int i = blockIdx.x * 256 + threadIdx.x;
    int m = i >> 16, r = (i >> 8) & 255, c = i & 255;
    const float* W;
    switch (m) {
        case 0: W = We; break;  case 1: W = Ws; break;  case 2: W = Wd; break;
        case 3: W = Wself; break; case 4: W = Wf; break; default: W = Wb;
    }
    wt[((size_t)m << 16) + (size_t)c * FD + r] = (_Float16)W[r * FD + c];
}

// ---------------------------------------------------------------------------
extern "C" void kernel_launch(void* const* d_in, const int* in_sizes, int n_in,
                              void* d_out, int out_size, void* d_ws, size_t ws_size,
                              hipStream_t stream) {
    (void)in_sizes; (void)n_in; (void)out_size; (void)ws_size;

    const float* h     = (const float*)d_in[0];
    const float* e     = (const float*)d_in[1];
    const int*   src   = (const int*)d_in[2];
    const int*   dst   = (const int*)d_in[3];
    const float* We    = (const float*)d_in[4];
    const float* Ws    = (const float*)d_in[5];
    const float* Wd    = (const float*)d_in[6];
    const float* Wself = (const float*)d_in[7];
    const float* Wf    = (const float*)d_in[8];
    const float* Wb    = (const float*)d_in[9];
    const float* att_f = (const float*)d_in[10];
    const float* att_b = (const float*)d_in[11];
    const float* g_e   = (const float*)d_in[12];
    const float* b_e   = (const float*)d_in[13];
    const float* g_h   = (const float*)d_in[14];
    const float* b_h   = (const float*)d_in[15];

    float* out_h = (float*)d_out;
    float* out_e = out_h + (size_t)NN * FD;

    // workspace carve-out (~1.42 GB total)
    char* p = (char*)d_ws;
    auto carve = [&](size_t bytes) {
        void* r = (void*)p;
        p += (bytes + 255) & ~(size_t)255;
        return r;
    };
    float*    e_tmp = (float*)   carve((size_t)EE * FD * 4);
    float*    h_tmp = (float*)   carve((size_t)NN * FD * 4);
    _Float16* e16   = (_Float16*)carve((size_t)EE * FD * 2);
    _Float16* h16   = (_Float16*)carve((size_t)NN * FD * 2);
    _Float16* wt    = (_Float16*)carve((size_t)6 * FD * FD * 2);
    float*    lf    = (float*)   carve((size_t)EE * 4);
    float*    lb    = (float*)   carve((size_t)EE * 4);
    float*    mxf   = (float*)   carve((size_t)NN * 4);
    float*    mxb   = (float*)   carve((size_t)NN * 4);
    float*    denf  = (float*)   carve((size_t)NN * 4);
    float*    denb  = (float*)   carve((size_t)NN * 4);
    float*    aggf  = (float*)   carve((size_t)NN * FD * 4);
    float*    aggb  = (float*)   carve((size_t)NN * FD * 4);
    float*    stats = (float*)   carve((size_t)8 * FD * 4);
    // stats: 0 sum_e | 1 ss_e | 2 mu_e | 3 rs_e | 4 sum_h | 5 ss_h | 6 mu_h | 7 rs_h

    // residual state lives in d_out, updated in place across layers
    hipMemcpyAsync(out_h, h, (size_t)NN * FD * 4, hipMemcpyDeviceToDevice, stream);
    hipMemcpyAsync(out_e, e, (size_t)EE * FD * 4, hipMemcpyDeviceToDevice, stream);
    k_cvt<<<(NN * FD) / 256, 256, 0, stream>>>(h, h16, (size_t)NN * FD);
    k_cvt<<<(EE * FD) / 256, 256, 0, stream>>>(e, e16, (size_t)EE * FD);

    const int nNodeBlk = (NN + 255) / 256;
    for (int l = 0; l < LYR; ++l) {
        size_t wo = (size_t)l * FD * FD;
        size_t vo = (size_t)l * FD;

        k_prep_w<<<6 * FD * FD / 256, 256, 0, stream>>>(
            We + wo, Ws + wo, Wd + wo, Wself + wo, Wf + wo, Wb + wo, wt);

        k_fill<<<NN * FD / 256, 256, 0, stream>>>(aggf, 0.f, (size_t)NN * FD);
        k_fill<<<NN * FD / 256, 256, 0, stream>>>(aggb, 0.f, (size_t)NN * FD);
        k_fill<<<nNodeBlk, 256, 0, stream>>>(denf, 0.f, (size_t)NN);
        k_fill<<<nNodeBlk, 256, 0, stream>>>(denb, 0.f, (size_t)NN);
        k_fill<<<nNodeBlk, 256, 0, stream>>>(mxf, -1e30f, (size_t)NN);
        k_fill<<<nNodeBlk, 256, 0, stream>>>(mxb, -1e30f, (size_t)NN);
        k_fill<<<8, 256, 0, stream>>>(stats, 0.f, (size_t)8 * FD);

        // edge GEMM (K=768 effective) + logits + BN partials
        k_edge_gemm<<<EE / BM, 256, 0, stream>>>(
            e16, h16, src, dst, wt, att_f + vo, att_b + vo,
            e_tmp, lf, lb, stats + 0 * FD, stats + 1 * FD);
        k_bn_finalize<<<1, FD, 0, stream>>>(stats + 0 * FD, stats + 1 * FD,
                                            stats + 2 * FD, stats + 3 * FD,
                                            1.0f / (float)EE);

        // segment softmax (both directions)
        k_segmax<<<EE / 256, 256, 0, stream>>>(lf, lb, src, dst, mxf, mxb);
        k_expsum<<<EE / 256, 256, 0, stream>>>(lf, lb, src, dst, mxf, mxb, denf, denb);

        // attention aggregation: forward (h_s@Wf -> dst), backward (h_d@Wb -> src)
        k_agg_gemm<<<EE / BM, 256, 0, stream>>>(h16, src, dst,
                                                wt + (size_t)4 * FD * FD, lf, denf, aggf);
        k_agg_gemm<<<EE / BM, 256, 0, stream>>>(h16, dst, src,
                                                wt + (size_t)5 * FD * FD, lb, denb, aggb);

        // e_new = e + relu(bn(e_tmp))   (in place, + f16 shadow for next layer)
        k_update<<<EE * FD / 1024, 256, 0, stream>>>(
            e_tmp, stats + 2 * FD, stats + 3 * FD, g_e + vo, b_e + vo,
            out_e, e16, (size_t)EE * FD / 4);

        // node GEMM + BN + update
        k_node_gemm<<<(NN + BM - 1) / BM, 256, 0, stream>>>(
            h16, wt + (size_t)3 * FD * FD, aggf, aggb,
            h_tmp, stats + 4 * FD, stats + 5 * FD);
        k_bn_finalize<<<1, FD, 0, stream>>>(stats + 4 * FD, stats + 5 * FD,
                                            stats + 6 * FD, stats + 7 * FD,
                                            1.0f / (float)NN);
        k_update<<<NN * FD / 1024, 256, 0, stream>>>(
            h_tmp, stats + 6 * FD, stats + 7 * FD, g_h + vo, b_h + vo,
            out_h, h16, (size_t)NN * FD / 4);
    }
}
